// SplineCNN8Residuals_78134045048904
// MI455X (gfx1250) — compile-verified, matrix operand order checked by reading, not run on hardware
//
#include <hip/hip_runtime.h>
#include <hip/hip_bf16.h>

#define NN 20000
#define NE 100000
#define NKERN 27
#define COLB 2336   // max packed K-columns per chunk (multiple of 32)

typedef __attribute__((ext_vector_type(16))) __bf16 v16bf;
typedef __attribute__((ext_vector_type(8)))  float  v8f;
typedef __attribute__((ext_vector_type(4))) unsigned int u32x4;
typedef __attribute__((ext_vector_type(8))) int i32x8;
typedef __attribute__((ext_vector_type(4))) int i32x4;

static inline int rup(int x, int m) { return (x + m - 1) / m * m; }
static inline long long cdivll(long long a, long long b) { return (a + b - 1) / b; }

__device__ inline unsigned short f2bf(float f) {
  unsigned int u = __float_as_uint(f);
  unsigned int r = (u + 0x7fffu + ((u >> 16) & 1u)) >> 16;
  return (unsigned short)r;
}

// ---------------- Tensor Data Mover staging (gfx1250) ----------------
#if defined(__gfx1250__) && __has_builtin(__builtin_amdgcn_tensor_load_to_lds) && \
    __has_builtin(__builtin_amdgcn_s_wait_tensorcnt)
#define USE_TDM 1
#endif

#ifdef USE_TDM
// Build a D# for a 2-D tile of 2-byte elements (tile0 x tile1) from a tensor
// with row stride `stride0` (elements), landing at LDS byte offset `lds_off`.
// Rows beyond td1 / cols beyond td0 read as zero (TDM OOB rule).
__device__ inline void tdm_load(unsigned lds_off, const void* gp,
                                unsigned td0, unsigned td1,
                                unsigned tile0, unsigned tile1, unsigned stride0) {
  unsigned long long ga = (unsigned long long)gp;
  u32x4 g0;
  g0[0] = 1u;                                                  // count=1 (valid, user)
  g0[1] = lds_off;                                             // lds_addr
  g0[2] = (unsigned)ga;                                        // global_addr[31:0]
  g0[3] = (unsigned)((ga >> 32) & 0x01FFFFFFu) | (2u << 30);   // addr[56:32] | type=2
  i32x8 g1;
  g1[0] = (int)(1u << 16);                                     // wg_mask=0, data_size=1 (2B)
  g1[1] = (int)((td0 & 0xFFFFu) << 16);                        // tensor_dim0[15:0]
  g1[2] = (int)(((td0 >> 16) & 0xFFFFu) | ((td1 & 0xFFFFu) << 16)); // dim0 hi | dim1 lo
  g1[3] = (int)(((td1 >> 16) & 0xFFFFu) | ((tile0 & 0xFFFFu) << 16)); // dim1 hi | tile0
  g1[4] = (int)(tile1 & 0xFFFFu);                              // tile1 | tile2=0
  g1[5] = (int)stride0;                                        // dim0_stride[31:0]
  g1[6] = 0;                                                   // stride hi | dim1_stride lo
  g1[7] = 0;
  i32x4 g2; g2[0] = 1; g2[1] = 0; g2[2] = 0; g2[3] = 0;        // tensor_dim2=1
  i32x4 g3; g3[0] = 0; g3[1] = 0; g3[2] = 0; g3[3] = 0;
#if __clang_major__ >= 23
  i32x8 g4 = {0, 0, 0, 0, 0, 0, 0, 0};
  __builtin_amdgcn_tensor_load_to_lds(g0, g1, g2, g3, g4, 0);
#else
  __builtin_amdgcn_tensor_load_to_lds(g0, g1, g2, g3, 0);
#endif
}
#endif

// ---------------- elementwise / setup kernels ----------------

__global__ void zero_kernel(float* p, long long n) {
  long long i = (long long)blockIdx.x * blockDim.x + threadIdx.x;
  if (i < n) p[i] = 0.f;
}

__global__ void basis_kernel(const float* __restrict__ ea, float* __restrict__ B) {
  int e = blockIdx.x * blockDim.x + threadIdx.x;
  if (e >= NE) return;
  float fr[3]; int lo[3];
#pragma unroll
  for (int d = 0; d < 3; d++) {
    float p = ea[e * 3 + d] * 3.0f;
    float fl = floorf(p);
    fr[d] = p - fl;
    lo[d] = (int)fl;
  }
  float acc[NKERN];
#pragma unroll
  for (int i = 0; i < NKERN; i++) acc[i] = 0.f;
#pragma unroll
  for (int j = 0; j < 8; j++) {
    int c0 = (j >> 2) & 1, c1 = (j >> 1) & 1, c2 = j & 1;
    float w = (c0 ? fr[0] : 1.f - fr[0]) * (c1 ? fr[1] : 1.f - fr[1]) * (c2 ? fr[2] : 1.f - fr[2]);
    int i0 = (lo[0] + c0) % 3, i1 = (lo[1] + c1) % 3, i2 = (lo[2] + c2) % 3;
    acc[i0 + 3 * i1 + 9 * i2] += w;
  }
#pragma unroll
  for (int i = 0; i < NKERN; i++) B[(long long)e * NKERN + i] = acc[i];
}

__global__ void degree_kernel(const int* __restrict__ dst, float* deg) {
  int e = blockIdx.x * blockDim.x + threadIdx.x;
  if (e >= NE) return;
  atomicAdd(&deg[dst[e]], 1.0f);
}

__global__ void invdeg_kernel(const float* deg, float* invdeg) {
  int n = blockIdx.x * blockDim.x + threadIdx.x;
  if (n < NN) invdeg[n] = 1.0f / fmaxf(deg[n], 1.0f);
}

// scatter: G[dst, (k-k0)*ci + c] += B[e,k] * X[src, c]
__global__ void scatter_kernel(const float* __restrict__ X, int ci,
                               const float* __restrict__ B,
                               const int* __restrict__ src, const int* __restrict__ dst,
                               float* __restrict__ G, int k0, int nk) {
  long long gid = (long long)blockIdx.x * blockDim.x + threadIdx.x;
  if (gid >= (long long)NE * ci) return;
  int e = (int)(gid / ci), c = (int)(gid % ci);
  int s = src[e], d = dst[e];
  float xv = X[(long long)s * ci + c];
  int wcols = nk * ci;
  float* gp = G + (long long)d * wcols + c;
  const float* bp = B + (long long)e * NKERN + k0;
  for (int k = 0; k < nk; k++) {
    float bw = bp[k];
    if (bw != 0.f) atomicAdd(gp + k * ci, bw * xv);
  }
}

__global__ void packA_kernel(const float* __restrict__ G, const float* __restrict__ invdeg,
                             int wcols, int kcpad, unsigned short* __restrict__ A) {
  long long gid = (long long)blockIdx.x * blockDim.x + threadIdx.x;
  if (gid >= (long long)NN * kcpad) return;
  int n = (int)(gid / kcpad), j = (int)(gid % kcpad);
  float v = 0.f;
  if (j < wcols) v = G[(long long)n * wcols + j] * invdeg[n];
  A[gid] = f2bf(v);
}

__global__ void packAroot_kernel(const float* __restrict__ X, int ci, int kcpad,
                                 unsigned short* __restrict__ A) {
  long long gid = (long long)blockIdx.x * blockDim.x + threadIdx.x;
  if (gid >= (long long)NN * kcpad) return;
  int n = (int)(gid / kcpad), j = (int)(gid % kcpad);
  float v = (j < ci) ? X[(long long)n * ci + j] : 0.f;
  A[gid] = f2bf(v);
}

// pack weights: layout Wb[kt][npad][32]; source w[k][c][co] row-major
__global__ void packW_kernel(const float* __restrict__ w, int ci, int co,
                             int k0, int wcols, int kcpad, int npad,
                             unsigned short* __restrict__ Wb) {
  long long gid = (long long)blockIdx.x * blockDim.x + threadIdx.x;
  if (gid >= (long long)kcpad * npad) return;
  int kk  = (int)(gid & 31);
  int col = (int)((gid >> 5) % npad);
  int kt  = (int)(gid / ((long long)npad * 32));
  int j = kt * 32 + kk;
  float v = 0.f;
  if (j < wcols && col < co) {
    int k = k0 + j / ci, c = j % ci;
    v = w[((long long)k * ci + c) * co + col];
  }
  Wb[gid] = f2bf(v);
}

__global__ void packWroot_kernel(const float* __restrict__ root, int ci, int co,
                                 int kcpad, int npad, unsigned short* __restrict__ Wb) {
  long long gid = (long long)blockIdx.x * blockDim.x + threadIdx.x;
  if (gid >= (long long)kcpad * npad) return;
  int kk  = (int)(gid & 31);
  int col = (int)((gid >> 5) % npad);
  int kt  = (int)(gid / ((long long)npad * 32));
  int j = kt * 32 + kk;
  float v = 0.f;
  if (j < ci && col < co) v = root[(long long)j * co + col];
  Wb[gid] = f2bf(v);
}

__global__ void initC_kernel(float* __restrict__ C, int npad,
                             const float* __restrict__ bias, int co) {
  long long gid = (long long)blockIdx.x * blockDim.x + threadIdx.x;
  if (gid >= (long long)NN * npad) return;
  int j = (int)(gid % npad);
  C[gid] = (j < co) ? bias[j] : 0.f;
}

// ---------------- WMMA GEMM with TDM->LDS double-buffered staging ----------------
// Block = 128 threads (4 waves). Wave w owns row tile rt = blockIdx.x*4 + w.
// Each wave holds NCT column accumulators; per K-tile: 1 A fragment, NCT WMMAs.
// A tile (64x32 bf16) and B tile (NCT*16 cols x 32 bf16) staged in LDS by TDM,
// double-buffered: load of tile t+1 overlaps compute of tile t.

template <int NCT>
__global__ void __launch_bounds__(128)
wmma_gemm_tile(const unsigned short* __restrict__ A,
               const unsigned short* __restrict__ Wb,
               float* __restrict__ C, int kcpad) {
  constexpr int A_ELEMS = 64 * 32;        // per buffer
  constexpr int B_ELEMS = NCT * 512;      // per buffer (NCT*16 cols x 32 k)
  constexpr int NPAD = NCT * 16;
  __shared__ unsigned short lds[2 * A_ELEMS + 2 * B_ELEMS];   // assumed at LDS offset 0

  const int tid = threadIdx.x;
  const int lane = tid & 31;
  const int w = tid >> 5;
  const int rowbase = blockIdx.x * 64;
  const int col = lane & 15;
  const int hk = (lane >> 4) << 3;        // 0 or 8
  const int ktiles = kcpad >> 5;

  v8f acc[NCT];
#pragma unroll
  for (int i = 0; i < NCT; i++) acc[i] = (v8f){};

  // ---- stage K-tile t into buffer bsel ----
  auto stage = [&](int t, int bsel) {
#ifdef USE_TDM
    if (tid < 32) {
      tdm_load((unsigned)(bsel * A_ELEMS * 2),
               A + (long long)rowbase * kcpad + (long long)t * 32,
               (unsigned)(kcpad - t * 32), (unsigned)(NN - rowbase),
               32u, 64u, (unsigned)kcpad);
      tdm_load((unsigned)((2 * A_ELEMS + bsel * B_ELEMS) * 2),
               Wb + (long long)t * B_ELEMS,
               (unsigned)B_ELEMS, 1u, (unsigned)B_ELEMS, 1u, (unsigned)B_ELEMS);
    }
#else
    for (int ch = tid; ch < 64 * 4; ch += 128) {       // A: 64 rows x 4 16B chunks
      int r = ch >> 2, p = (ch & 3) * 8;
      *(uint4*)(lds + bsel * A_ELEMS + r * 32 + p) =
          *(const uint4*)(A + (long long)(rowbase + r) * kcpad + (long long)t * 32 + p);
    }
    for (int ch = tid; ch < B_ELEMS / 8; ch += 128) {  // B: contiguous
      *(uint4*)(lds + 2 * A_ELEMS + bsel * B_ELEMS + ch * 8) =
          *(const uint4*)(Wb + (long long)t * B_ELEMS + ch * 8);
    }
#endif
  };
  auto wait_stage = [&]() {
#ifdef USE_TDM
    if (tid < 32) __builtin_amdgcn_s_wait_tensorcnt(0);
#endif
  };

  stage(0, 0);
  wait_stage();
  __syncthreads();

  for (int t = 0; t < ktiles; ++t) {
    const int cur = t & 1;
    if (t + 1 < ktiles) stage(t + 1, cur ^ 1);   // overlap next load with compute

    union { v16bf v; uint4 q[2]; } fa;
    const unsigned short* ap = lds + cur * A_ELEMS + (w * 16 + col) * 32 + hk;
    fa.q[0] = *(const uint4*)ap;
    fa.q[1] = *(const uint4*)(ap + 16);
    const unsigned short* bbase = lds + 2 * A_ELEMS + cur * B_ELEMS + hk;
#pragma unroll
    for (int ctl = 0; ctl < NCT; ++ctl) {
      union { v16bf v; uint4 q[2]; } fb;
      const unsigned short* bp = bbase + (ctl * 16 + col) * 32;
      fb.q[0] = *(const uint4*)bp;
      fb.q[1] = *(const uint4*)(bp + 16);
      acc[ctl] = __builtin_amdgcn_wmma_f32_16x16x32_bf16(false, fa.v, false, fb.v,
                                                         (short)0, acc[ctl], false, false);
    }
    wait_stage();
    __syncthreads();
  }

  const int rt = blockIdx.x * 4 + w;
  if (rt < NN / 16) {
    float* cp = C + ((long long)rt * 16 + (lane >> 4) * 8) * NPAD + col;
#pragma unroll
    for (int ctl = 0; ctl < NCT; ++ctl) {
#pragma unroll
      for (int r = 0; r < 8; r++) cp[(long long)r * NPAD + ctl * 16] += acc[ctl][r];
    }
  }
}

// ---------------- BN / ELU / copies ----------------

__global__ void bn_stats_kernel(const float* __restrict__ X, int stride,
                                float* mu, float* rsig) {
  int f = blockIdx.x;
  __shared__ float s1[256], s2[256];
  float a = 0.f, b = 0.f;
  for (int n = threadIdx.x; n < NN; n += 256) {
    float v = X[(long long)n * stride + f];
    a += v; b += v * v;
  }
  s1[threadIdx.x] = a; s2[threadIdx.x] = b;
  __syncthreads();
  for (int s = 128; s > 0; s >>= 1) {
    if (threadIdx.x < s) { s1[threadIdx.x] += s1[threadIdx.x + s]; s2[threadIdx.x] += s2[threadIdx.x + s]; }
    __syncthreads();
  }
  if (threadIdx.x == 0) {
    float m = s1[0] / (float)NN;
    float var = s2[0] / (float)NN - m * m;
    mu[f] = m;
    rsig[f] = rsqrtf(var + 1e-5f);
  }
}

__global__ void bn_elu_kernel(const float* __restrict__ X, int stride, int F,
                              const float* __restrict__ g, const float* __restrict__ b,
                              const float* __restrict__ mu, const float* __restrict__ rsig,
                              float* __restrict__ Y) {
  long long gid = (long long)blockIdx.x * blockDim.x + threadIdx.x;
  if (gid >= (long long)NN * F) return;
  int n = (int)(gid / F), f = (int)(gid % F);
  float v = g[f] * (X[(long long)n * stride + f] - mu[f]) * rsig[f] + b[f];
  Y[gid] = (v > 0.f) ? v : expm1f(v);
}

__global__ void copy_cols_kernel(const float* __restrict__ X, int xstride, int F,
                                 float* __restrict__ Y, int ystride, int yoff) {
  long long gid = (long long)blockIdx.x * blockDim.x + threadIdx.x;
  if (gid >= (long long)NN * F) return;
  int n = (int)(gid / F), f = (int)(gid % F);
  Y[(long long)n * ystride + yoff + f] = X[(long long)n * xstride + f];
}

// ---------------- host orchestration ----------------

struct Ctx {
  const int* src; const int* dst;
  const float* Bb; const float* invdeg;
  float* G; unsigned short* Ab; unsigned short* Wb; float* C;
  float* mu; float* rsig;
  hipStream_t s;
};

static void launch_gemm(Ctx& c, const unsigned short* A, int kcpad, int npad) {
  dim3 grid((NN / 16 + 3) / 4), block(128);
  switch (npad / 16) {
    case 1:  wmma_gemm_tile<1><<<grid, block, 0, c.s>>>(A, c.Wb, c.C, kcpad); break;
    case 2:  wmma_gemm_tile<2><<<grid, block, 0, c.s>>>(A, c.Wb, c.C, kcpad); break;
    case 4:  wmma_gemm_tile<4><<<grid, block, 0, c.s>>>(A, c.Wb, c.C, kcpad); break;
    case 8:  wmma_gemm_tile<8><<<grid, block, 0, c.s>>>(A, c.Wb, c.C, kcpad); break;
    default: wmma_gemm_tile<16><<<grid, block, 0, c.s>>>(A, c.Wb, c.C, kcpad); break;
  }
}

static void run_conv(Ctx& c, const float* X, int ci, int co,
                     const float* w, const float* root, const float* bias, int& npad_out) {
  int npad = rup(co, 16);
  long long ctot = (long long)NN * npad;
  initC_kernel<<<(int)cdivll(ctot, 256), 256, 0, c.s>>>(c.C, npad, bias, co);

  int maxk = NKERN;
  if (maxk * ci > COLB) maxk = COLB / ci;
  for (int k0 = 0; k0 < NKERN; k0 += maxk) {
    int nk = (NKERN - k0 < maxk) ? (NKERN - k0) : maxk;
    int wcols = nk * ci;
    int kcpad = rup(wcols, 32);
    long long gtot = (long long)NN * wcols;
    zero_kernel<<<(int)cdivll(gtot, 256), 256, 0, c.s>>>(c.G, gtot);
    long long stot = (long long)NE * ci;
    scatter_kernel<<<(int)cdivll(stot, 256), 256, 0, c.s>>>(X, ci, c.Bb, c.src, c.dst, c.G, k0, nk);
    long long atot = (long long)NN * kcpad;
    packA_kernel<<<(int)cdivll(atot, 256), 256, 0, c.s>>>(c.G, c.invdeg, wcols, kcpad, c.Ab);
    long long wtot = (long long)kcpad * npad;
    packW_kernel<<<(int)cdivll(wtot, 256), 256, 0, c.s>>>(w, ci, co, k0, wcols, kcpad, npad, c.Wb);
    launch_gemm(c, c.Ab, kcpad, npad);
  }
  // root linear term as an extra K-chunk (unscaled by inv_deg)
  {
    int kcpad = rup(ci, 32);
    long long atot = (long long)NN * kcpad;
    packAroot_kernel<<<(int)cdivll(atot, 256), 256, 0, c.s>>>(X, ci, kcpad, c.Ab);
    long long wtot = (long long)kcpad * npad;
    packWroot_kernel<<<(int)cdivll(wtot, 256), 256, 0, c.s>>>(root, ci, co, kcpad, npad, c.Wb);
    launch_gemm(c, c.Ab, kcpad, npad);
  }
  npad_out = npad;
}

static void run_block(Ctx& c, const float* resin, int ci, int co, int mid,
                      const float* const* P, const float* x,
                      float* r1b, float* r2b, float* catb, float* outb) {
  int npad;
  // conv1 -> BN -> ELU
  run_conv(c, resin, ci, mid, P[0], P[1], P[2], npad);
  bn_stats_kernel<<<mid, 256, 0, c.s>>>(c.C, npad, c.mu, c.rsig);
  long long t1 = (long long)NN * mid;
  bn_elu_kernel<<<(int)cdivll(t1, 256), 256, 0, c.s>>>(c.C, npad, mid, P[9], P[10], c.mu, c.rsig, r1b);
  // conv2 -> BN -> ELU
  int mid2 = 2 * mid;
  run_conv(c, r1b, mid, mid2, P[3], P[4], P[5], npad);
  bn_stats_kernel<<<mid2, 256, 0, c.s>>>(c.C, npad, c.mu, c.rsig);
  long long t2 = (long long)NN * mid2;
  bn_elu_kernel<<<(int)cdivll(t2, 256), 256, 0, c.s>>>(c.C, npad, mid2, P[11], P[12], c.mu, c.rsig, r2b);
  // cat = [r2 | x]
  int ci3 = mid2 + 3;
  copy_cols_kernel<<<(int)cdivll(t2, 256), 256, 0, c.s>>>(r2b, mid2, mid2, catb, ci3, 0);
  copy_cols_kernel<<<(int)cdivll((long long)NN * 3, 256), 256, 0, c.s>>>(x, 3, 3, catb, ci3, mid2);
  // conv3 -> compact output
  run_conv(c, catb, ci3, co, P[6], P[7], P[8], npad);
  long long t3 = (long long)NN * co;
  copy_cols_kernel<<<(int)cdivll(t3, 256), 256, 0, c.s>>>(c.C, npad, co, outb, co, 0);
}

extern "C" void kernel_launch(void* const* d_in, const int* in_sizes, int n_in,
                              void* d_out, int out_size, void* d_ws, size_t ws_size,
                              hipStream_t stream) {
  const float* x  = (const float*)d_in[0];
  const int*   ei = (const int*)d_in[1];
  const float* ea = (const float*)d_in[2];
  const float* P[8][13];
  int idx = 3;
  for (int b = 0; b < 8; b++)
    for (int j = 0; j < 13; j++) P[b][j] = (const float*)d_in[idx++];

  // carve workspace (256B aligned slices)
  char* base = (char*)d_ws;
  auto take = [&](size_t bytes) -> char* {
    char* p = base;
    base += (bytes + 255) / 256 * 256;
    return p;
  };
  float*          Bb     = (float*)take((size_t)NE * NKERN * 4);
  float*          deg    = (float*)take((size_t)NN * 4);
  float*          invdeg = (float*)take((size_t)NN * 4);
  float*          mu     = (float*)take(256 * 4);
  float*          rsig   = (float*)take(256 * 4);
  float*          G      = (float*)take((size_t)NN * COLB * 4);
  unsigned short* Ab     = (unsigned short*)take((size_t)NN * COLB * 2);
  unsigned short* Wb     = (unsigned short*)take((size_t)COLB * 256 * 2);
  float*          C      = (float*)take((size_t)NN * 256 * 4);
  float*          r1b    = (float*)take((size_t)NN * 128 * 4);
  float*          r2b    = (float*)take((size_t)NN * 256 * 4);
  float*          catb   = (float*)take((size_t)NN * 259 * 4);
  float*          resA   = (float*)take((size_t)NN * 64 * 4);
  float*          resB   = (float*)take((size_t)NN * 64 * 4);
  float*          res4   = (float*)take((size_t)NN * 8 * 4);
  float*          binb   = (float*)take((size_t)NN * 75 * 4);

  const int* src = ei;
  const int* dst = ei + NE;

  basis_kernel<<<(int)cdivll(NE, 256), 256, 0, stream>>>(ea, Bb);
  zero_kernel<<<(int)cdivll(NN, 256), 256, 0, stream>>>(deg, NN);
  degree_kernel<<<(int)cdivll(NE, 256), 256, 0, stream>>>(dst, deg);
  invdeg_kernel<<<(int)cdivll(NN, 256), 256, 0, stream>>>(deg, invdeg);

  Ctx c{src, dst, Bb, invdeg, G, Ab, Wb, C, mu, rsig, stream};

  run_block(c, x,    3, 16,  8,  P[0], x, r1b, r2b, catb, resA);
  run_block(c, resA, 16, 64, 32,  P[1], x, r1b, r2b, catb, resB);
  run_block(c, resB, 64, 64, 128, P[2], x, r1b, r2b, catb, resA);
  run_block(c, resA, 64, 8,  16,  P[3], x, r1b, r2b, catb, res4);
  // bin = [res4 | x] (11)
  copy_cols_kernel<<<(int)cdivll((long long)NN * 8, 256), 256, 0, stream>>>(res4, 8, 8, binb, 11, 0);
  copy_cols_kernel<<<(int)cdivll((long long)NN * 3, 256), 256, 0, stream>>>(x, 3, 3, binb, 11, 8);
  run_block(c, binb, 11, 32, 16,  P[4], x, r1b, r2b, catb, resA);
  run_block(c, resA, 32, 64, 32,  P[5], x, r1b, r2b, catb, resB);
  run_block(c, resB, 64, 64, 128, P[6], x, r1b, r2b, catb, resA);
  // bin = [resA | res4 | x] (75)
  copy_cols_kernel<<<(int)cdivll((long long)NN * 64, 256), 256, 0, stream>>>(resA, 64, 64, binb, 75, 0);
  copy_cols_kernel<<<(int)cdivll((long long)NN * 8, 256), 256, 0, stream>>>(res4, 8, 8, binb, 75, 64);
  copy_cols_kernel<<<(int)cdivll((long long)NN * 3, 256), 256, 0, stream>>>(x, 3, 3, binb, 75, 72);
  // final block writes compact [NN,4] straight into d_out
  run_block(c, binb, 75, 4, 16,  P[7], x, r1b, r2b, catb, (float*)d_out);
}
